// WindowAttention_29068338660125
// MI455X (gfx1250) — matrix operand order repaired and checked by reading, not compiled, not run on hardware
//
#include <hip/hip_runtime.h>
#include <math.h>

#define NTOK 64
#define DIM  256
#define NH   8
#define HD   32
#define C3   768

#define LDX 264   // x / out-staging LDS row stride (bf16 elems)
#define LDQ 776   // qkv LDS row stride (bf16 elems)
#define LDA 68    // attn scratch row stride (f32 elems)
#define LDV 72    // transposed-V row stride (bf16 elems)

#define XS_OFF  0
#define QKV_OFF 33792
#define SQ_OFF  133120
#define AT_OFF  137216
#define VT_OFF  276480
#define SMEM_BYTES 313344   // <= 320KB per WGP

typedef __attribute__((ext_vector_type(16))) __bf16 v16bf;
typedef __attribute__((ext_vector_type(8)))  float  v8f;

union FragAB { uint4 q[2]; v16bf v; unsigned short h[16]; };
union FragC  { v8f v; float f[8]; };

__device__ __forceinline__ unsigned short f2bf(float f) {
  unsigned int u = __float_as_uint(f);
  u += 0x7fffu + ((u >> 16) & 1u);        // round-to-nearest-even
  return (unsigned short)(u >> 16);
}
__device__ __forceinline__ float bf2f(unsigned short h) {
  return __uint_as_float(((unsigned int)h) << 16);
}
__device__ __forceinline__ v8f wmma_bf16(const FragAB& a, const FragAB& b, v8f c) {
  return __builtin_amdgcn_wmma_f32_16x16x32_bf16(false, a.v, false, b.v,
                                                 (short)0, c, false, false);
}

// ---------------------------------------------------------------------------
// Pack qkv_w (256x768) and proj_w (256x256) into WMMA B-fragment order, bf16.
// Fragment (nt,kt): 512 bf16; lane L owns elems [L*16, L*16+16): K = 8*(L/16)+(j&7)+16*(j>>3),
// N = nt*16 + (L%16).  => per-lane fragment load is one 32B contiguous chunk.
// ---------------------------------------------------------------------------
__global__ void pack_weights(const float* __restrict__ qkv_w,
                             const float* __restrict__ proj_w,
                             unsigned short* __restrict__ qkv_wp,
                             unsigned short* __restrict__ proj_wp) {
  int idx = blockIdx.x * blockDim.x + threadIdx.x;
  const int NQKV = 48 * 8 * 512;          // 196608
  const int NPRJ = 16 * 8 * 512;          // 65536
  if (idx >= NQKV + NPRJ) return;
  if (idx < NQKV) {
    int frag = idx >> 9, e = idx & 511;
    int nt = frag >> 3, kt = frag & 7;
    int lane = e >> 4, j = e & 15;
    int n = nt * 16 + (lane & 15);
    int k = kt * 32 + (lane >> 4) * 8 + (j & 7) + (j >> 3) * 16;
    qkv_wp[idx] = f2bf(qkv_w[k * C3 + n]);
  } else {
    int p = idx - NQKV;
    int frag = p >> 9, e = p & 511;
    int nt = frag >> 3, kt = frag & 7;
    int lane = e >> 4, j = e & 15;
    int n = nt * 16 + (lane & 15);
    int k = kt * 32 + (lane >> 4) * 8 + (j & 7) + (j >> 3) * 16;
    proj_wp[p] = f2bf(proj_w[k * DIM + n]);
  }
}

// ---------------------------------------------------------------------------
// CPB MLP -> gathered 16*sigmoid bias (8,64,64), per-head scales, fused qkv bias
// ---------------------------------------------------------------------------
__global__ void cpb_bias(const float* __restrict__ w1, const float* __restrict__ b1,
                         const float* __restrict__ w2, const float* __restrict__ ls,
                         const float* __restrict__ q_bias, const float* __restrict__ v_bias,
                         float* __restrict__ bias_g, float* __restrict__ scales_g,
                         float* __restrict__ qkvb_g) {
  __shared__ float tbl[225 * 8];
  int tid = threadIdx.x;
  for (int t = tid; t < 225 * 8; t += 256) {
    int r = t >> 3, h = t & 7;
    int i = r / 15, j = r % 15;
    float v0 = (float)(i - 7) * (8.0f / 7.0f);
    float v1 = (float)(j - 7) * (8.0f / 7.0f);
    float c0 = (v0 >= 0.f ? 1.f : -1.f) * log2f(fabsf(v0) + 1.f) * (1.f / 3.f);
    float c1 = (v1 >= 0.f ? 1.f : -1.f) * log2f(fabsf(v1) + 1.f) * (1.f / 3.f);
    float acc = 0.f;
    for (int jj = 0; jj < 512; ++jj) {
      float hv = fmaxf(c0 * w1[jj] + c1 * w1[512 + jj] + b1[jj], 0.f);
      acc += hv * w2[jj * 8 + h];
    }
    tbl[t] = acc;
  }
  __syncthreads();
  for (int e = tid; e < NH * NTOK * NTOK; e += 256) {
    int h = e >> 12, m = (e >> 6) & 63, n = e & 63;
    int dr = (m >> 3) - (n >> 3) + 7;
    int dc = (m & 7) - (n & 7) + 7;
    float xv = tbl[(dr * 15 + dc) * 8 + h];
    bias_g[e] = 16.f / (1.f + __expf(-xv));
  }
  for (int e = tid; e < C3; e += 256)
    qkvb_g[e] = (e < 256) ? q_bias[e] : ((e < 512) ? 0.f : v_bias[e - 512]);
  if (tid < NH) scales_g[tid] = __expf(fminf(ls[tid], 4.6051702f)); // log(100)
}

// ---------------------------------------------------------------------------
// Fused per-window attention kernel: one block = one window (64 tokens).
// ---------------------------------------------------------------------------
__launch_bounds__(256, 1)
__global__ void fused_attn(const float* __restrict__ x,
                           const unsigned short* __restrict__ qkv_wp,
                           const unsigned short* __restrict__ proj_wp,
                           const float* __restrict__ qkvb_g,
                           const float* __restrict__ proj_b,
                           const float* __restrict__ bias_g,
                           const float* __restrict__ scales_g,
                           float* __restrict__ out) {
  extern __shared__ __align__(16) char smem[];
  unsigned short* xs   = (unsigned short*)(smem + XS_OFF);   // 64 x LDX bf16 (x, later out-stage)
  unsigned short* qkvs = (unsigned short*)(smem + QKV_OFF);  // 64 x LDQ bf16
  float*          sq   = (float*)(smem + SQ_OFF);            // [2][64][8] rsqrt norms
  float*          attnf= (float*)(smem + AT_OFF);            // 8 x 64 x LDA f32 (also x fp32 stage)
  unsigned short* vt   = (unsigned short*)(smem + VT_OFF);   // 256 x LDV bf16 (V transposed)

  const int tid  = threadIdx.x;
  const int wave = tid >> 5;
  const int lane = tid & 31;
  const int lm   = lane & 15;
  const int lh   = lane >> 4;
  const int b    = blockIdx.x;

  // ---- Phase 1: async-copy x tile (fp32) into LDS, then convert to bf16 ---
  const float* xb = x + (size_t)b * NTOK * DIM;
  #pragma unroll
  for (int i = 0; i < 16; ++i) {
    int fidx = tid + i * 256;                        // 4096 x 16B
    unsigned ldsoff = (unsigned)(AT_OFF + fidx * 16);
    const float4* gp = ((const float4*)xb) + fidx;
    asm volatile("global_load_async_to_lds_b128 %0, %1, off"
                 :: "v"(ldsoff), "v"(gp) : "memory");
  }
  asm volatile("s_wait_asynccnt 0x0" ::: "memory");
  __syncthreads();
  const float* xf = (const float*)(smem + AT_OFF);
  #pragma unroll
  for (int i = 0; i < 16; ++i) {
    int fidx = tid + i * 256;
    float4 t4 = ((const float4*)xf)[fidx];
    int e = fidx << 2;
    int row = e >> 8, col = e & 255;
    uint2 pk;
    pk.x = (unsigned)f2bf(t4.x) | ((unsigned)f2bf(t4.y) << 16);
    pk.y = (unsigned)f2bf(t4.z) | ((unsigned)f2bf(t4.w) << 16);
    *(uint2*)(xs + row * LDX + col) = pk;
  }
  __syncthreads();

  // ---- Phase 2: QKV GEMM, 2x2 register blocking --------------------------
  for (int t = wave; t < 48; t += 8) {
    int bm = t / 24, bn = t % 24;                    // 2x2 tile block
    int mt0 = bm * 2, nt0 = bn * 2;
    FragC a00, a01, a10, a11;
    #pragma unroll
    for (int r = 0; r < 8; ++r) { a00.f[r]=0.f; a01.f[r]=0.f; a10.f[r]=0.f; a11.f[r]=0.f; }
    const unsigned short* wp0 = qkv_wp + ((nt0 * 8) << 9) + (lane << 4);
    const unsigned short* wp1 = wp0 + (8 << 9);
    const unsigned short* ar0 = xs + (mt0 * 16 + lm) * LDX;
    const unsigned short* ar1 = ar0 + 16 * LDX;
    #pragma unroll
    for (int kt = 0; kt < 8; ++kt) {
      FragAB fa0, fa1, fb0, fb1;
      fa0.q[0] = *(const uint4*)(ar0 + kt * 32 + lh * 8);
      fa0.q[1] = *(const uint4*)(ar0 + kt * 32 + 16 + lh * 8);
      fa1.q[0] = *(const uint4*)(ar1 + kt * 32 + lh * 8);
      fa1.q[1] = *(const uint4*)(ar1 + kt * 32 + 16 + lh * 8);
      fb0.q[0] = ((const uint4*)wp0)[0];
      fb0.q[1] = ((const uint4*)wp0)[1];
      fb1.q[0] = ((const uint4*)wp1)[0];
      fb1.q[1] = ((const uint4*)wp1)[1];
      if (kt < 7) { __builtin_prefetch(wp0 + 512, 0, 1); __builtin_prefetch(wp1 + 512, 0, 1); }
      wp0 += 512; wp1 += 512;
      a00.v = wmma_bf16(fa0, fb0, a00.v);
      a01.v = wmma_bf16(fa0, fb1, a01.v);
      a10.v = wmma_bf16(fa1, fb0, a10.v);
      a11.v = wmma_bf16(fa1, fb1, a11.v);
    }
    #pragma unroll
    for (int jm = 0; jm < 2; ++jm) {
      #pragma unroll
      for (int jn = 0; jn < 2; ++jn) {
        FragC* ac = jm ? (jn ? &a11 : &a10) : (jn ? &a01 : &a00);
        int mt = mt0 + jm, nt = nt0 + jn;
        int col = nt * 16 + lm;
        float bv = qkvb_g[col];
        #pragma unroll
        for (int r = 0; r < 8; ++r) {
          int row = mt * 16 + r + lh * 8;
          unsigned short hv = f2bf(ac->f[r] + bv);
          qkvs[row * LDQ + col] = hv;
          if (nt >= 32) vt[(col - 512) * LDV + row] = hv;   // transposed V copy
        }
      }
    }
  }
  __syncthreads();

  // ---- per-(token,head) reciprocal L2 norms of q and k --------------------
  #pragma unroll
  for (int i = 0; i < 4; ++i) {
    int task = tid + i * 256;                 // 1024 rows of 32
    int qk = task >> 9, rem = task & 511;
    int tok = rem >> 3, h = rem & 7;
    const unsigned short* p = qkvs + tok * LDQ + qk * 256 + h * 32;
    float s = 0.f;
    #pragma unroll
    for (int j = 0; j < 32; ++j) { float v = bf2f(p[j]); s += v * v; }
    sq[task] = rsqrtf(fmaxf(s, 1e-12f));
  }
  __syncthreads();

  // ---- Phase 3: attention, one head per wave ------------------------------
  {
    const int h = wave;
    const float scale_h = scales_g[h];
    float* ab = attnf + h * NTOK * LDA;

    // logits = (q . k) * rq * rk * scale + bias   (2x2 blocked, K=32 = head dim)
    #pragma unroll
    for (int bm = 0; bm < 2; ++bm) {
      #pragma unroll
      for (int bn = 0; bn < 2; ++bn) {
        FragAB qa0, qa1, kb0, kb1;
        const unsigned short* q0 = qkvs + ((bm * 2) * 16 + lm) * LDQ + h * 32;
        const unsigned short* q1 = q0 + 16 * LDQ;
        qa0.q[0] = *(const uint4*)(q0 + lh * 8);
        qa0.q[1] = *(const uint4*)(q0 + 16 + lh * 8);
        qa1.q[0] = *(const uint4*)(q1 + lh * 8);
        qa1.q[1] = *(const uint4*)(q1 + 16 + lh * 8);
        const unsigned short* k0 = qkvs + ((bn * 2) * 16 + lm) * LDQ + 256 + h * 32 + lh * 8;
        const unsigned short* k1 = k0 + 16 * LDQ;
        kb0.q[0] = ((const uint4*)k0)[0];
        kb0.q[1] = ((const uint4*)k0)[2];
        kb1.q[0] = ((const uint4*)k1)[0];
        kb1.q[1] = ((const uint4*)k1)[2];
        FragC c00, c01, c10, c11;
        #pragma unroll
        for (int r = 0; r < 8; ++r) { c00.f[r]=0.f; c01.f[r]=0.f; c10.f[r]=0.f; c11.f[r]=0.f; }
        c00.v = wmma_bf16(qa0, kb0, c00.v);
        c01.v = wmma_bf16(qa0, kb1, c01.v);
        c10.v = wmma_bf16(qa1, kb0, c10.v);
        c11.v = wmma_bf16(qa1, kb1, c11.v);
        #pragma unroll
        for (int jm = 0; jm < 2; ++jm) {
          #pragma unroll
          for (int jn = 0; jn < 2; ++jn) {
            FragC* ac = jm ? (jn ? &c11 : &c10) : (jn ? &c01 : &c00);
            int mt = bm * 2 + jm, nt = bn * 2 + jn;
            int n = nt * 16 + lm;
            float rk = sq[512 + n * 8 + h];
            #pragma unroll
            for (int r = 0; r < 8; ++r) {
              int m = mt * 16 + r + lh * 8;
              float rq = sq[m * 8 + h];
              ab[m * LDA + n] = ac->f[r] * rq * rk * scale_h + bias_g[(h * 64 + m) * 64 + n];
            }
          }
        }
      }
    }

    // softmax: each lane owns rows (lane) and (lane+32) entirely in registers
    #pragma unroll
    for (int rep = 0; rep < 2; ++rep) {
      int m = lane + rep * 32;
      float* rp = ab + m * LDA;
      float v[64];
      #pragma unroll
      for (int j4 = 0; j4 < 16; ++j4) {
        float4 t4 = ((const float4*)rp)[j4];
        v[j4 * 4 + 0] = t4.x; v[j4 * 4 + 1] = t4.y;
        v[j4 * 4 + 2] = t4.z; v[j4 * 4 + 3] = t4.w;
      }
      float mx = v[0];
      #pragma unroll
      for (int j = 1; j < 64; ++j) mx = fmaxf(mx, v[j]);
      float s = 0.f;
      #pragma unroll
      for (int j = 0; j < 64; ++j) { float e = __expf(v[j] - mx); v[j] = e; s += e; }
      float inv = __frcp_rn(s);
      unsigned short* rh = (unsigned short*)rp;   // probs bf16, in-place (row is lane-private)
      #pragma unroll
      for (int j = 0; j < 64; ++j) rh[j] = f2bf(v[j] * inv);
    }

    // out_h = P(64x64) @ V_h(64x32); B from transposed V (contiguous loads)
    #pragma unroll
    for (int bm = 0; bm < 2; ++bm) {
      #pragma unroll
      for (int nt = 0; nt < 2; ++nt) {
        FragC c0, c1;
        #pragma unroll
        for (int r = 0; r < 8; ++r) { c0.f[r] = 0.f; c1.f[r] = 0.f; }
        int d = h * 32 + nt * 16 + lm;              // flat transposed-V row
        #pragma unroll
        for (int ks = 0; ks < 2; ++ks) {
          FragAB pa0, pa1, vb;
          const unsigned short* p0 = (const unsigned short*)(ab + ((bm * 2) * 16 + lm) * LDA);
          const unsigned short* p1 = (const unsigned short*)(ab + ((bm * 2) * 16 + 16 + lm) * LDA);
          pa0.q[0] = *(const uint4*)(p0 + ks * 32 + lh * 8);
          pa0.q[1] = *(const uint4*)(p0 + ks * 32 + 16 + lh * 8);
          pa1.q[0] = *(const uint4*)(p1 + ks * 32 + lh * 8);
          pa1.q[1] = *(const uint4*)(p1 + ks * 32 + 16 + lh * 8);
          const unsigned short* vp = vt + d * LDV + ks * 32 + lh * 8;
          vb.q[0] = ((const uint4*)vp)[0];
          vb.q[1] = ((const uint4*)vp)[2];
          c0.v = wmma_bf16(pa0, vb, c0.v);
          c1.v = wmma_bf16(pa1, vb, c1.v);
        }
        #pragma unroll
        for (int jm = 0; jm < 2; ++jm) {
          FragC* ac = jm ? &c1 : &c0;
          #pragma unroll
          for (int r = 0; r < 8; ++r) {           // stage into xs (x no longer needed)
            int row = (bm * 2 + jm) * 16 + r + lh * 8;
            xs[row * LDX + h * 32 + nt * 16 + lm] = f2bf(ac->f[r]);
          }
        }
      }
    }
  }
  __syncthreads();

  // ---- Phase 4: projection, 2x2 register blocking -------------------------
  float* ob = out + (size_t)b * NTOK * DIM;
  for (int t = wave; t < 16; t += 8) {
    int bm = t / 8, bn = t % 8;
    int mt0 = bm * 2, nt0 = bn * 2;
    FragC a00, a01, a10, a11;
    #pragma unroll
    for (int r = 0; r < 8; ++r) { a00.f[r]=0.f; a01.f[r]=0.f; a10.f[r]=0.f; a11.f[r]=0.f; }
    const unsigned short* wp0 = proj_wp + ((nt0 * 8) << 9) + (lane << 4);
    const unsigned short* wp1 = wp0 + (8 << 9);
    const unsigned short* ar0 = xs + (mt0 * 16 + lm) * LDX;
    const unsigned short* ar1 = ar0 + 16 * LDX;
    #pragma unroll
    for (int kt = 0; kt < 8; ++kt) {
      FragAB fa0, fa1, fb0, fb1;
      fa0.q[0] = *(const uint4*)(ar0 + kt * 32 + lh * 8);
      fa0.q[1] = *(const uint4*)(ar0 + kt * 32 + 16 + lh * 8);
      fa1.q[0] = *(const uint4*)(ar1 + kt * 32 + lh * 8);
      fa1.q[1] = *(const uint4*)(ar1 + kt * 32 + 16 + lh * 8);
      fb0.q[0] = ((const uint4*)wp0)[0];
      fb0.q[1] = ((const uint4*)wp0)[1];
      fb1.q[0] = ((const uint4*)wp1)[0];
      fb1.q[1] = ((const uint4*)wp1)[1];
      if (kt < 7) { __builtin_prefetch(wp0 + 512, 0, 1); __builtin_prefetch(wp1 + 512, 0, 1); }
      wp0 += 512; wp1 += 512;
      a00.v = wmma_bf16(fa0, fb0, a00.v);
      a01.v = wmma_bf16(fa0, fb1, a01.v);
      a10.v = wmma_bf16(fa1, fb0, a10.v);
      a11.v = wmma_bf16(fa1, fb1, a11.v);
    }
    #pragma unroll
    for (int jm = 0; jm < 2; ++jm) {
      #pragma unroll
      for (int jn = 0; jn < 2; ++jn) {
        FragC* ac = jm ? (jn ? &a11 : &a10) : (jn ? &a01 : &a00);
        int mt = mt0 + jm, nt = nt0 + jn;
        int col = nt * 16 + lm;
        float pb = proj_b[col];
        #pragma unroll
        for (int r = 0; r < 8; ++r) {
          int row = mt * 16 + r + lh * 8;
          ob[row * DIM + col] = ac->f[r] + pb;
        }
      }
    }
  }
}

// ---------------------------------------------------------------------------
extern "C" void kernel_launch(void* const* d_in, const int* in_sizes, int n_in,
                              void* d_out, int out_size, void* d_ws, size_t ws_size,
                              hipStream_t stream) {
  (void)in_sizes; (void)n_in; (void)out_size; (void)ws_size;
  const float* x           = (const float*)d_in[0];
  const float* qkv_w       = (const float*)d_in[1];
  const float* q_bias      = (const float*)d_in[2];
  const float* v_bias      = (const float*)d_in[3];
  const float* logit_scale = (const float*)d_in[4];
  const float* cpb_w1      = (const float*)d_in[5];
  const float* cpb_b1      = (const float*)d_in[6];
  const float* cpb_w2      = (const float*)d_in[7];
  const float* proj_w      = (const float*)d_in[8];
  const float* proj_b      = (const float*)d_in[9];
  float* out = (float*)d_out;

  char* ws = (char*)d_ws;
  unsigned short* qkv_wp  = (unsigned short*)(ws + 0);       // 393216 B
  unsigned short* proj_wp = (unsigned short*)(ws + 393216);  // 131072 B
  float*          bias_g  = (float*)(ws + 524288);           // 131072 B
  float*          scales_g= (float*)(ws + 655360);           // 32 B
  float*          qkvb_g  = (float*)(ws + 655392);           // 3072 B

  pack_weights<<<1024, 256, 0, stream>>>(qkv_w, proj_w, qkv_wp, proj_wp);
  cpb_bias<<<1, 256, 0, stream>>>(cpb_w1, cpb_b1, cpb_w2, logit_scale,
                                  q_bias, v_bias, bias_g, scales_g, qkvb_g);

  hipFuncSetAttribute(reinterpret_cast<const void*>(fused_attn),
                      hipFuncAttributeMaxDynamicSharedMemorySize, SMEM_BYTES);
  fused_attn<<<4096, 256, SMEM_BYTES, stream>>>(x, qkv_wp, proj_wp, qkvb_g,
                                                proj_b, bias_g, scales_g, out);
}
